// PointConvW_51170240364927
// MI455X (gfx1250) — compile-verified, matrix operand order checked by reading, not run on hardware
//
#include <hip/hip_runtime.h>
#include <math.h>

#define B_ 8
#define N_ 8192
#define D_ 64
#define O_ 128
#define S_ 2048
#define K_ 16
#define C_ (D_ + 3)   // 67
#define CP 96         // C padded to 3 x 32 for WMMA K-steps
#define LEAKY_F 0.1f

typedef __attribute__((ext_vector_type(16))) _Float16 v16h;
typedef __attribute__((ext_vector_type(8)))  float    v8f;
typedef __attribute__((ext_vector_type(4)))  unsigned int u32x4;
typedef __attribute__((ext_vector_type(8)))  int          i32x8;
typedef __attribute__((ext_vector_type(4)))  int          i32x4;

__device__ __forceinline__ float leaky(float x) { return x >= 0.f ? x : LEAKY_F * x; }

// ---------------------------------------------------------------------------
// TDM: async-DMA one [3 x 1024] f32 tile (row stride N_) from global into LDS.
// D# group0: count=1, lds_addr, global_addr, type=2. group1: data_size=4B,
// tensor_dim0=1024 (tile never OOB), tensor_dim1=3, tile_dim0=1024,
// tile_dim1=3, tensor_dim0_stride=8192. Tracked by TENSORcnt (per-wave,
// in-order), EXEC-independent; issued from wave 0 only.
// ---------------------------------------------------------------------------
__device__ __forceinline__ void tdm_load_tile3x1024(const float* gsrc,
                                                    unsigned lds_addr) {
  unsigned long long ga = (unsigned long long)(size_t)gsrc;
  u32x4 g0;
  g0[0] = 1u;                                   // count=1, user mode
  g0[1] = lds_addr;                             // bits 63:32 lds_addr
  g0[2] = (unsigned)ga;                         // global_addr lo
  g0[3] = (unsigned)((ga >> 32) & 0x01FFFFFFu) | (2u << 30);  // addr hi | type=2
  i32x8 g1;
  g1[0] = (2 << 16);            // wg_mask=0, data_size=2 (4B)
  g1[1] = (1024 << 16);         // tensor_dim0[15:0] @ bits63:48
  g1[2] = (3 << 16);            // tensor_dim0 hi=0 | tensor_dim1 lo=3
  g1[3] = (1024 << 16);         // tensor_dim1 hi=0 | tile_dim0=1024 @127:112
  g1[4] = 3;                    // tile_dim1=3, tile_dim2=0
  g1[5] = 8192;                 // tensor_dim0_stride lo
  g1[6] = 0;                    // stride0 hi=0, stride1 lo=0 (2D tile: unused)
  g1[7] = 0;
  i32x4 g2; g2[0] = 0; g2[1] = 0; g2[2] = 0; g2[3] = 0;  // dims 2/3 unused
  i32x4 g3 = g2;
#if __clang_major__ >= 23
  i32x8 g4;
  g4[0]=0; g4[1]=0; g4[2]=0; g4[3]=0; g4[4]=0; g4[5]=0; g4[6]=0; g4[7]=0;
  __builtin_amdgcn_tensor_load_to_lds(g0, g1, g2, g3, g4, 0);
#else
  __builtin_amdgcn_tensor_load_to_lds(g0, g1, g2, g3, 0);
#endif
}

// ---------------------------------------------------------------------------
// Kernel 1: furthest point sampling. One block (1024 thr = 32 waves) per batch.
// All xyz + dist state lives in registers (8 points/thread); argmax via
// wave shfl_xor reduction + 32-entry LDS tree. Serial-latency bound.
// ---------------------------------------------------------------------------
__global__ __launch_bounds__(1024) void fps_kernel(
    const float* __restrict__ xyz,
    float* __restrict__ out_nx,    // [B,3,S]
    float* __restrict__ out_fps,   // [B,S] (values as float)
    float* __restrict__ ws_nxyz,   // [B,S,3]
    int*   __restrict__ ws_fps)    // [B,S]
{
  const int b    = blockIdx.x;
  const int tid  = threadIdx.x;
  const int lane = tid & 31, wave = tid >> 5;
  const float* xb = xyz + (size_t)b * 3 * N_;

  float px[8], py[8], pz[8], dist[8];
#pragma unroll
  for (int j = 0; j < 8; ++j) {
    int p = tid + j * 1024;
    px[j] = xb[p];
    py[j] = xb[N_ + p];
    pz[j] = xb[2 * N_ + p];
    dist[j] = 1e10f;
  }

  __shared__ float sC[3];
  __shared__ float sWd[32];
  __shared__ int   sWp[32];
  __shared__ int   sFar;

  int far = 0;
  for (int i = 0; i < S_; ++i) {
    if (tid == (far & 1023)) {
      int j = far >> 10;
      sC[0] = px[j]; sC[1] = py[j]; sC[2] = pz[j];
      out_nx[(size_t)b * 3 * S_ + i]            = px[j];
      out_nx[(size_t)b * 3 * S_ + S_ + i]       = py[j];
      out_nx[(size_t)b * 3 * S_ + 2 * S_ + i]   = pz[j];
      ws_nxyz[((size_t)b * S_ + i) * 3 + 0] = px[j];
      ws_nxyz[((size_t)b * S_ + i) * 3 + 1] = py[j];
      ws_nxyz[((size_t)b * S_ + i) * 3 + 2] = pz[j];
    }
    if (tid == 0) { ws_fps[b * S_ + i] = far; out_fps[b * S_ + i] = (float)far; }
    __syncthreads();

    float cx = sC[0], cy = sC[1], cz = sC[2];
    float bd = -1.f; int bp = 0x7fffffff;
#pragma unroll
    for (int j = 0; j < 8; ++j) {
      float dx = px[j] - cx, dy = py[j] - cy, dz = pz[j] - cz;
      float d = dx * dx + dy * dy + dz * dz;
      dist[j] = fminf(dist[j], d);
      int p = tid + j * 1024;
      if (dist[j] > bd || (dist[j] == bd && p < bp)) { bd = dist[j]; bp = p; }
    }
#pragma unroll
    for (int off = 16; off > 0; off >>= 1) {
      float od = __shfl_xor(bd, off, 32);
      int   op = __shfl_xor(bp, off, 32);
      if (od > bd || (od == bd && op < bp)) { bd = od; bp = op; }
    }
    if (lane == 0) { sWd[wave] = bd; sWp[wave] = bp; }
    __syncthreads();
    if (wave == 0) {
      bd = sWd[lane]; bp = sWp[lane];
#pragma unroll
      for (int off = 16; off > 0; off >>= 1) {
        float od = __shfl_xor(bd, off, 32);
        int   op = __shfl_xor(bp, off, 32);
        if (od > bd || (od == bd && op < bp)) { bd = od; bp = op; }
      }
      if (lane == 0) sFar = bp;
    }
    __syncthreads();
    far = sFar;
  }
}

// ---------------------------------------------------------------------------
// Kernel 2: 16-NN. One thread per query. Candidate xyz tiles are DMA'd by the
// Tensor Data Mover into a double-buffered LDS region: wave 0 issues tile T+1,
// s_wait_tensorcnt(1) proves tile T landed (per-wave in-order completion),
// barrier releases the block to scan tile T while T+1 streams in.
// Register-resident sorted top-16 insertion (stable on ties = top_k order).
// ---------------------------------------------------------------------------
__global__ __launch_bounds__(256) void knn_kernel(
    const float* __restrict__ xyz,
    const float* __restrict__ ws_nxyz,
    int* __restrict__ ws_knn)            // [B,S,K]
{
  const int b    = blockIdx.y;
  const int tid  = threadIdx.x;
  const int wave = tid >> 5;
  const int s    = blockIdx.x * 256 + tid;
  const float* xb = xyz + (size_t)b * 3 * N_;

  float qx = ws_nxyz[((size_t)b * S_ + s) * 3 + 0];
  float qy = ws_nxyz[((size_t)b * S_ + s) * 3 + 1];
  float qz = ws_nxyz[((size_t)b * S_ + s) * 3 + 2];

  __shared__ float sBuf[2][3 * 1024];    // 24KB double buffer (x|y|z rows)

  float bd[16]; int bi[16];
#pragma unroll
  for (int t = 0; t < 16; ++t) { bd[t] = 3.4e38f; bi[t] = 0; }

  if (wave == 0) {
    tdm_load_tile3x1024(xb, (unsigned)(size_t)&sBuf[0][0]);
  }

  for (int T = 0; T < N_ / 1024; ++T) {
    if (wave == 0) {
      if (T < N_ / 1024 - 1) {
        tdm_load_tile3x1024(xb + (T + 1) * 1024,
                            (unsigned)(size_t)&sBuf[(T + 1) & 1][0]);
        __builtin_amdgcn_s_wait_tensorcnt(1);   // tile T complete
      } else {
        __builtin_amdgcn_s_wait_tensorcnt(0);
      }
    }
    __syncthreads();                            // tile T visible to all waves
    const float* bx = &sBuf[T & 1][0];
    const float* by = bx + 1024;
    const float* bz = bx + 2048;
    for (int u = 0; u < 1024; ++u) {
      float dx = bx[u] - qx, dy = by[u] - qy, dz = bz[u] - qz;
      float d = dx * dx + dy * dy + dz * dz;
      if (d < bd[15]) {
        bd[15] = d; bi[15] = T * 1024 + u;
#pragma unroll
        for (int t = 15; t > 0; --t) {
          if (bd[t] < bd[t - 1]) {
            float td = bd[t]; bd[t] = bd[t - 1]; bd[t - 1] = td;
            int   ti = bi[t]; bi[t] = bi[t - 1]; bi[t - 1] = ti;
          }
        }
      }
    }
    __syncthreads();                            // done reading before reuse
  }
#pragma unroll
  for (int t = 0; t < 16; ++t) ws_knn[((size_t)b * S_ + s) * K_ + t] = bi[t];
}

// ---------------------------------------------------------------------------
// Kernel 3: fused gather + WMMA feat GEMM + leaky + all aggregations/gating.
// Block = 256 thr (8 waves); each wave owns one query s and computes the
// full feat[:,s,:] = [128 x 16] tile in 8 v8f WMMA accumulators
// (8 M-tiles x 3 K-steps of v_wmma_f32_16x16x32_f16). Everything downstream
// (ch/pt means, 144x144 + 16x16 + 128x128 matvecs, sigmoid gates, gated
// k-mean) stays in the wave => feat never touches HBM (saves ~268 MB).
// ---------------------------------------------------------------------------
__global__ __launch_bounds__(256) void feat_kernel(
    const float* __restrict__ xyz, const float* __restrict__ points,
    const float* __restrict__ Wk, const float* __restrict__ Wl,
    const float* __restrict__ Wpt, const float* __restrict__ Wc,
    const float* __restrict__ ws_nxyz, const int* __restrict__ ws_knn,
    float* __restrict__ out)             // [B,O,S]
{
  const int b    = blockIdx.y;
  const int tid  = threadIdx.x;
  const int lane = tid & 31, wave = tid >> 5;
  const int s    = blockIdx.x * 8 + wave;

  __shared__ _Float16 sWk[O_ * CP];   // 24 KB, f16 padded W_kernel
  __shared__ float sAgg[8][160];      // per-wave agg (144)
  __shared__ float sAgg2[8][160];     // per-wave leaky(W_linear @ agg)
  __shared__ float sWpv[8][16];       // per-wave point gate
  __shared__ float sWcv[8][128];      // per-wave channel gate

  // Warm L2/WGP$ for the later matvec weights while we do WMMA work.
  __builtin_prefetch(Wl + (size_t)tid * 81, 0, 0);
  __builtin_prefetch(Wc + (size_t)tid * 64, 0, 0);

  for (int e = tid; e < O_ * CP; e += 256) {
    int o = e / CP, c = e % CP;
    sWk[e] = (_Float16)(c < C_ ? Wk[o * C_ + c] : 0.f);
  }
  __syncthreads();

  const int nk   = lane & 15;   // WMMA N index = neighbor k
  const int half = lane >> 4;
  const int gidx = ws_knn[((size_t)b * S_ + s) * K_ + nk];
  const float q0 = ws_nxyz[((size_t)b * S_ + s) * 3 + 0];
  const float q1 = ws_nxyz[((size_t)b * S_ + s) * 3 + 1];
  const float q2 = ws_nxyz[((size_t)b * S_ + s) * 3 + 2];
  const float* xb = xyz + (size_t)b * 3 * N_;
  const float* pb = points + (size_t)b * D_ * N_;

  // B fragments: column n = k-neighbor, rows = channels (ISA 16-bit K layout)
  v16h Bt[3];
#pragma unroll
  for (int t = 0; t < 3; ++t) {
#pragma unroll
    for (int h = 0; h < 16; ++h) {
      int c = t * 32 + (h < 8 ? h : h + 8) + half * 8;
      float v;
      if (c < 3) {
        float pv = (c == 0 ? xb[gidx] : (c == 1 ? xb[N_ + gidx] : xb[2 * N_ + gidx]));
        v = pv - (c == 0 ? q0 : (c == 1 ? q1 : q2));
      } else if (c < C_) {
        v = pb[(size_t)(c - 3) * N_ + gidx];
      } else {
        v = 0.f;
      }
      Bt[t][h] = (_Float16)v;
    }
  }

  v8f zero = {0.f, 0.f, 0.f, 0.f, 0.f, 0.f, 0.f, 0.f};
  v8f acc[8];
#pragma unroll
  for (int m = 0; m < 8; ++m) acc[m] = zero;

#pragma unroll
  for (int m = 0; m < 8; ++m) {
    int o = 16 * m + nk;
#pragma unroll
    for (int t = 0; t < 3; ++t) {
      v16h A;
#pragma unroll
      for (int h = 0; h < 16; ++h) {
        int c = t * 32 + (h < 8 ? h : h + 8) + half * 8;
        A[h] = sWk[o * CP + c];
      }
      acc[m] = __builtin_amdgcn_wmma_f32_16x16x32_f16(
          false, A, false, Bt[t], (short)0, acc[m], false, false);
    }
  }

  // leaky relu on feat tile
#pragma unroll
  for (int m = 0; m < 8; ++m)
#pragma unroll
    for (int r = 0; r < 8; ++r) acc[m][r] = leaky(acc[m][r]);

  // ch_avg[o]: reduce over k (cross-lane within 16-lane half-group)
#pragma unroll
  for (int m = 0; m < 8; ++m) {
#pragma unroll
    for (int r = 0; r < 8; ++r) {
      float v = acc[m][r];
      v += __shfl_xor(v, 1, 32); v += __shfl_xor(v, 2, 32);
      v += __shfl_xor(v, 4, 32); v += __shfl_xor(v, 8, 32);
      if (nk == 0) sAgg[wave][16 * m + r + 8 * half] = v * (1.f / 16.f);
    }
  }
  // pt_avg[k]: reduce over o (sum VGPR rows + cross-half shfl)
  {
    float pt = 0.f;
#pragma unroll
    for (int m = 0; m < 8; ++m)
#pragma unroll
      for (int r = 0; r < 8; ++r) pt += acc[m][r];
    pt += __shfl_xor(pt, 16, 32);
    if (lane < 16) sAgg[wave][128 + lane] = pt * (1.f / 128.f);
  }
  // Same-wave LDS ops are in-order (DScnt) -> no barrier needed (wave-private).

  // agg2 = leaky(W_linear[144x144] @ agg)
#pragma unroll
  for (int rr0 = 0; rr0 < 5; ++rr0) {
    int rr = lane + rr0 * 32;
    if (rr < 144) {
      float a = 0.f;
      for (int c = 0; c < 144; ++c) a += Wl[rr * 144 + c] * sAgg[wave][c];
      sAgg2[wave][rr] = leaky(a);
    }
  }
  // point gate (16x16) and channel gate (128x128)
  if (lane < 16) {
    float a = 0.f;
#pragma unroll
    for (int c = 0; c < 16; ++c) a += Wpt[lane * 16 + c] * sAgg2[wave][128 + c];
    sWpv[wave][lane] = 1.f / (1.f + __expf(-a));
  }
#pragma unroll
  for (int rr0 = 0; rr0 < 4; ++rr0) {
    int rr = lane + rr0 * 32;
    float a = 0.f;
    for (int c = 0; c < 128; ++c) a += Wc[rr * 128 + c] * sAgg2[wave][c];
    sWcv[wave][rr] = 1.f / (1.f + __expf(-a));
  }

  // out[o,s] = wc[o]/16 * sum_k feat[o,k] * wp[k]
  float wpk = sWpv[wave][nk];
#pragma unroll
  for (int m = 0; m < 8; ++m) {
#pragma unroll
    for (int r = 0; r < 8; ++r) {
      float v = acc[m][r] * wpk;
      v += __shfl_xor(v, 1, 32); v += __shfl_xor(v, 2, 32);
      v += __shfl_xor(v, 4, 32); v += __shfl_xor(v, 8, 32);
      if (nk == 0) {
        int o = 16 * m + r + 8 * half;
        out[(size_t)b * O_ * S_ + (size_t)o * S_ + s] = v * sWcv[wave][o] * (1.f / 16.f);
      }
    }
  }
}

// ---------------------------------------------------------------------------
extern "C" void kernel_launch(void* const* d_in, const int* in_sizes, int n_in,
                              void* d_out, int out_size, void* d_ws, size_t ws_size,
                              hipStream_t stream) {
  (void)in_sizes; (void)n_in; (void)out_size; (void)ws_size;
  const float* xyz    = (const float*)d_in[0];
  const float* points = (const float*)d_in[1];
  const float* Wk     = (const float*)d_in[2];
  const float* Wl     = (const float*)d_in[3];
  const float* Wpt    = (const float*)d_in[4];
  const float* Wc     = (const float*)d_in[5];

  float* out      = (float*)d_out;
  float* out_nx   = out;                                   // [B,3,S]
  float* out_feat = out + (size_t)B_ * 3 * S_;             // [B,O,S]
  float* out_fps  = out + (size_t)B_ * 3 * S_ + (size_t)B_ * O_ * S_;  // [B,S]

  char* ws = (char*)d_ws;
  float* ws_nxyz = (float*)ws;                                               // B*S*3 f32
  int*   ws_fps  = (int*)(ws + (size_t)B_ * S_ * 3 * sizeof(float));         // B*S i32
  int*   ws_knn  = (int*)(ws + (size_t)B_ * S_ * 3 * sizeof(float)
                             + (size_t)B_ * S_ * sizeof(int));               // B*S*K i32

  fps_kernel<<<B_, 1024, 0, stream>>>(xyz, out_nx, out_fps, ws_nxyz, ws_fps);
  knn_kernel<<<dim3(S_ / 256, B_), 256, 0, stream>>>(xyz, ws_nxyz, ws_knn);
  feat_kernel<<<dim3(S_ / 8, B_), 256, 0, stream>>>(xyz, points, Wk, Wl, Wpt, Wc,
                                                    ws_nxyz, ws_knn, out_feat);
}
